// BiRNNBowman_55868934586821
// MI455X (gfx1250) — matrix operand order repaired and checked by reading, not compiled, not run on hardware
//
#include <hip/hip_runtime.h>
#include <hip/hip_bf16.h>
#include <math.h>

// ---------------------------------------------------------------------------
// Model dims
// ---------------------------------------------------------------------------
#define BB 256     // batch
#define TT 128     // time steps
#define EE 300     // embedding
#define EP 320     // embedding padded to 32-multiple (10 k-tiles)
#define HH 512     // hidden
#define KP 832     // padded K for step GEMM: 320 (x) + 512 (h)
#define N4H 2048   // 4*H
#define FF 1024    // MLP width

typedef __attribute__((ext_vector_type(16))) __bf16 v16bf;
typedef __attribute__((ext_vector_type(8)))  __bf16 bf16x8;
typedef __attribute__((ext_vector_type(8)))  float  v8f;

__device__ __forceinline__ __bf16 f2bf(float f) {
    unsigned u = __builtin_bit_cast(unsigned, f);
    unsigned r = u + 0x7FFFu + ((u >> 16) & 1u);      // round-to-nearest-even
    unsigned short h = (unsigned short)(r >> 16);
    return __builtin_bit_cast(__bf16, h);
}
__device__ __forceinline__ float bf2f(__bf16 x) {
    unsigned short h = __builtin_bit_cast(unsigned short, x);
    unsigned u = ((unsigned)h) << 16;
    return __builtin_bit_cast(float, u);
}
__device__ __forceinline__ float sigm(float x) { return 1.0f / (1.0f + expf(-x)); }

// A-fragment (16x32 bf16): lane L holds row m0+(L&15), K = {h*16..h*16+15}
__device__ __forceinline__ v16bf load_afrag(const __bf16* __restrict__ row, int half) {
    bf16x8 lo = *(const bf16x8*)(row + half * 16);
    bf16x8 hi = *(const bf16x8*)(row + half * 16 + 8);
    v16bf a;
    #pragma unroll
    for (int i = 0; i < 8; ++i) { a[i] = lo[i]; a[8 + i] = hi[i]; }
    return a;
}

// ---------------------------------------------------------------------------
// Setup / conversion kernels
// ---------------------------------------------------------------------------

// xp[2][B][T][EP] bf16, zero-padded cols 300..319
__global__ void k_convert_x(const float* __restrict__ prem, const float* __restrict__ hyp,
                            __bf16* __restrict__ xp) {
    long idx = (long)blockIdx.x * blockDim.x + threadIdx.x;
    const long total = 2L * BB * TT * EP;
    if (idx >= total) return;
    int k = (int)(idx % EP);
    long r = idx / EP;
    int t = (int)(r % TT);
    long rb = r / TT;
    int b = (int)(rb % BB);
    int i = (int)(rb / BB);
    float v = 0.0f;
    if (k < EE) {
        const float* src = (i == 0) ? prem : hyp;
        v = src[((long)b * TT + t) * EE + k];
    }
    xp[idx] = f2bf(v);
}

// Wtp[4][N4H][KP] bf16 transposed+padded: rows 0..299 -> k 0..299, rows 300..811 -> k 320..831
__global__ void k_convert_lstm_w(const float* __restrict__ W0, const float* __restrict__ W1,
                                 const float* __restrict__ W2, const float* __restrict__ W3,
                                 __bf16* __restrict__ Wtp) {
    long idx = (long)blockIdx.x * blockDim.x + threadIdx.x;
    const long total = 4L * N4H * KP;
    if (idx >= total) return;
    int k = (int)(idx % KP);
    long r = idx / KP;
    int n = (int)(r % N4H);
    int l = (int)(r / N4H);
    const float* W = (l == 0) ? W0 : (l == 1) ? W1 : (l == 2) ? W2 : W3;
    float v = 0.0f;
    if (k < EE)        v = W[(long)k * N4H + n];
    else if (k >= EP)  v = W[(long)(k - (EP - EE)) * N4H + n];   // row k-20
    Wtp[idx] = f2bf(v);
}

__global__ void k_copy_bias(const float* __restrict__ b0, const float* __restrict__ b1,
                            const float* __restrict__ b2, const float* __restrict__ b3,
                            float* __restrict__ bias_all) {
    int idx = blockIdx.x * blockDim.x + threadIdx.x;
    if (idx >= 4 * N4H) return;
    int l = idx >> 11, j = idx & (N4H - 1);
    const float* s = (l == 0) ? b0 : (l == 1) ? b1 : (l == 2) ? b2 : b3;
    bias_all[idx] = s[j];
}

// Wt[n][k] = W[k][n]  (generic fp32 -> bf16 transpose)
__global__ void k_convert_mlp_w(const float* __restrict__ W, __bf16* __restrict__ Wt,
                                int N, int K) {
    long idx = (long)blockIdx.x * blockDim.x + threadIdx.x;
    if (idx >= (long)N * K) return;
    int k = (int)(idx % K);
    int n = (int)(idx / K);
    Wt[idx] = f2bf(W[(long)k * N + n]);
}

__global__ void k_init_hc(__bf16* __restrict__ h, float* __restrict__ c) {
    int idx = blockIdx.x * blockDim.x + threadIdx.x;
    if (idx >= 4 * BB * HH) return;
    h[idx] = f2bf(0.0f);
    c[idx] = 0.0f;
}

// ---------------------------------------------------------------------------
// LSTM step: z[l] = [x_t | h_l] @ W_l   for all 4 LSTMs  (bf16 WMMA, f32 accum)
// 2048 waves, each computes a 32x32 output tile (2x2 register blocking):
// per k-tile 8 loads feed 4 WMMAs.
// ---------------------------------------------------------------------------
__global__ void lstm_step_gemm(const __bf16* __restrict__ xp,   // [2][B][T][EP]
                               const __bf16* __restrict__ Wtp,  // [4][N4H][KP]
                               const __bf16* __restrict__ h,    // [4][B][H]
                               float* __restrict__ z,           // [4][B][N4H]
                               int s) {
    int gtid = blockIdx.x * blockDim.x + threadIdx.x;
    int w    = gtid >> 5;
    int lane = gtid & 31;
    int nblk = w & 63;           // 2048 / 32
    int mblk = (w >> 6) & 7;     // 256 / 32
    int l    = w >> 9;           // LSTM index (512 waves per LSTM)
    int m0 = mblk << 5, n0 = nblk << 5;
    int half = lane >> 4, lm = lane & 15;

    int inp = l >> 1;                         // premises / hypotheses
    int t   = (l & 1) ? (TT - 1 - s) : s;     // bw LSTMs read reversed time

    const __bf16* Ax0 = xp + (((long)inp * BB + m0 + lm)      * TT + t) * EP;
    const __bf16* Ax1 = xp + (((long)inp * BB + m0 + 16 + lm) * TT + t) * EP;
    const __bf16* Ah0 = h  + ((long)l * BB + m0 + lm)      * HH;
    const __bf16* Ah1 = h  + ((long)l * BB + m0 + 16 + lm) * HH;
    const __bf16* Bw0 = Wtp + (long)l * N4H * KP + (long)(n0 + lm) * KP;
    const __bf16* Bw1 = Bw0 + 16L * KP;

    v8f acc00 = {}, acc01 = {}, acc10 = {}, acc11 = {};
    #pragma unroll 2
    for (int kt = 0; kt < KP / 32; ++kt) {
        int k0 = kt << 5;
        bool xpart = (kt < EP / 32);
        const __bf16* a0p = xpart ? (Ax0 + k0) : (Ah0 + (k0 - EP));
        const __bf16* a1p = xpart ? (Ax1 + k0) : (Ah1 + (k0 - EP));
        v16bf a0 = load_afrag(a0p, half);
        v16bf a1 = load_afrag(a1p, half);
        v16bf b0 = *(const v16bf*)(Bw0 + k0 + half * 16);
        v16bf b1 = *(const v16bf*)(Bw1 + k0 + half * 16);
        acc00 = __builtin_amdgcn_wmma_f32_16x16x32_bf16(false, a0, false, b0, (short)0, acc00, false, false);
        acc01 = __builtin_amdgcn_wmma_f32_16x16x32_bf16(false, a0, false, b1, (short)0, acc01, false, false);
        acc10 = __builtin_amdgcn_wmma_f32_16x16x32_bf16(false, a1, false, b0, (short)0, acc10, false, false);
        acc11 = __builtin_amdgcn_wmma_f32_16x16x32_bf16(false, a1, false, b1, (short)0, acc11, false, false);
    }
    float* zl = z + (long)l * BB * N4H;
    #pragma unroll
    for (int v = 0; v < 8; ++v) {
        int mA = m0 + half * 8 + v;
        int mB = mA + 16;
        zl[(long)mA * N4H + n0 + lm]      = acc00[v];
        zl[(long)mA * N4H + n0 + 16 + lm] = acc01[v];
        zl[(long)mB * N4H + n0 + lm]      = acc10[v];
        zl[(long)mB * N4H + n0 + 16 + lm] = acc11[v];
    }
}

// Gate math: c = c*sig(f+1) + sig(i)*tanh(j); h = tanh(c)*sig(o)
__global__ void lstm_cell(const float* __restrict__ z, const float* __restrict__ bias_all,
                          float* __restrict__ c, __bf16* __restrict__ h) {
    int idx = blockIdx.x * blockDim.x + threadIdx.x;
    if (idx >= 4 * BB * HH) return;
    int l = idx >> 17;              // 256*512 = 2^17
    int r = idx & (BB * HH - 1);
    int b = r >> 9, u = r & (HH - 1);
    const float* zr = z + ((long)l * BB + b) * N4H;
    const float* bb = bias_all + l * N4H;
    float gi = zr[u]            + bb[u];
    float gj = zr[HH + u]       + bb[HH + u];
    float gf = zr[2 * HH + u]   + bb[2 * HH + u];
    float go = zr[3 * HH + u]   + bb[3 * HH + u];
    float cc = c[idx];
    cc = cc * sigm(gf + 1.0f) + sigm(gi) * tanhf(gj);
    c[idx] = cc;
    h[idx] = f2bf(tanhf(cc) * sigm(go));
}

// rnn_out[b][l*512+u] = c[l][b][u]  (concat order fw_p, bw_p, fw_h, bw_h)
__global__ void k_repack(const float* __restrict__ c, __bf16* __restrict__ rnnbf) {
    int idx = blockIdx.x * blockDim.x + threadIdx.x;
    if (idx >= 4 * BB * HH) return;
    int l = idx >> 17;
    int r = idx & (BB * HH - 1);
    int b = r >> 9, u = r & (HH - 1);
    rnnbf[(long)b * N4H + l * HH + u] = f2bf(c[idx]);
}

// ---------------------------------------------------------------------------
// MLP layer: out = tanh(A @ Wt^T + bias), bf16 WMMA, 2x2 blocking, M=256 fixed
// ---------------------------------------------------------------------------
__global__ void mlp_gemm_tanh(const __bf16* __restrict__ A, const __bf16* __restrict__ Wt,
                              const float* __restrict__ bias, __bf16* __restrict__ out,
                              int N, int K) {
    int gtid = blockIdx.x * blockDim.x + threadIdx.x;
    int w    = gtid >> 5;
    int lane = gtid & 31;
    int Nb = N >> 5;             // n-blocks of 32
    int nblk = w % Nb;
    int mblk = w / Nb;
    int m0 = mblk << 5, n0 = nblk << 5;
    int half = lane >> 4, lm = lane & 15;

    const __bf16* a0row = A  + (long)(m0 + lm)      * K;
    const __bf16* a1row = A  + (long)(m0 + 16 + lm) * K;
    const __bf16* b0row = Wt + (long)(n0 + lm)      * K;
    const __bf16* b1row = b0row + 16L * K;

    v8f acc00 = {}, acc01 = {}, acc10 = {}, acc11 = {};
    for (int kt = 0; kt < (K >> 5); ++kt) {
        int k0 = kt << 5;
        v16bf a0 = load_afrag(a0row + k0, half);
        v16bf a1 = load_afrag(a1row + k0, half);
        v16bf b0 = *(const v16bf*)(b0row + k0 + half * 16);
        v16bf b1 = *(const v16bf*)(b1row + k0 + half * 16);
        acc00 = __builtin_amdgcn_wmma_f32_16x16x32_bf16(false, a0, false, b0, (short)0, acc00, false, false);
        acc01 = __builtin_amdgcn_wmma_f32_16x16x32_bf16(false, a0, false, b1, (short)0, acc01, false, false);
        acc10 = __builtin_amdgcn_wmma_f32_16x16x32_bf16(false, a1, false, b0, (short)0, acc10, false, false);
        acc11 = __builtin_amdgcn_wmma_f32_16x16x32_bf16(false, a1, false, b1, (short)0, acc11, false, false);
    }
    float bc0 = bias[n0 + lm];
    float bc1 = bias[n0 + 16 + lm];
    #pragma unroll
    for (int v = 0; v < 8; ++v) {
        int mA = m0 + half * 8 + v;
        int mB = mA + 16;
        out[(long)mA * N + n0 + lm]      = f2bf(tanhf(acc00[v] + bc0));
        out[(long)mA * N + n0 + 16 + lm] = f2bf(tanhf(acc01[v] + bc1));
        out[(long)mB * N + n0 + lm]      = f2bf(tanhf(acc10[v] + bc0));
        out[(long)mB * N + n0 + 16 + lm] = f2bf(tanhf(acc11[v] + bc1));
    }
}

// logits = a3 @ W4 + b4  (N=3, plain fp32)
__global__ void k_logits(const __bf16* __restrict__ a3, const float* __restrict__ W4,
                         const float* __restrict__ b4, float* __restrict__ out) {
    int idx = blockIdx.x * blockDim.x + threadIdx.x;
    if (idx >= BB * 3) return;
    int b = idx / 3, j = idx % 3;
    float acc = b4[j];
    const __bf16* ar = a3 + (long)b * FF;
    for (int k = 0; k < FF; ++k) acc += bf2f(ar[k]) * W4[(long)k * 3 + j];
    out[idx] = acc;
}

// ---------------------------------------------------------------------------
// Host
// ---------------------------------------------------------------------------
extern "C" void kernel_launch(void* const* d_in, const int* in_sizes, int n_in,
                              void* d_out, int out_size, void* d_ws, size_t ws_size,
                              hipStream_t stream) {
    const float* premises = (const float*)d_in[0];
    const float* hyps     = (const float*)d_in[1];
    const float* W_lstm[4] = { (const float*)d_in[2], (const float*)d_in[4],
                               (const float*)d_in[6], (const float*)d_in[8] };
    const float* b_lstm[4] = { (const float*)d_in[3], (const float*)d_in[5],
                               (const float*)d_in[7], (const float*)d_in[9] };
    const float* W1 = (const float*)d_in[10]; const float* b1 = (const float*)d_in[11];
    const float* W2 = (const float*)d_in[12]; const float* b2 = (const float*)d_in[13];
    const float* W3 = (const float*)d_in[14]; const float* b3 = (const float*)d_in[15];
    const float* W4 = (const float*)d_in[16]; const float* b4 = (const float*)d_in[17];
    float* out = (float*)d_out;

    char* ws = (char*)d_ws;
    size_t off = 0;
    auto alloc = [&](size_t bytes) -> void* {
        off = (off + 255) & ~(size_t)255;
        void* p = ws + off;
        off += bytes;
        return p;
    };
    __bf16* xp      = (__bf16*)alloc(2L * BB * TT * EP * 2);
    __bf16* Wtp     = (__bf16*)alloc(4L * N4H * KP * 2);
    float*  biasAll = (float*) alloc(4L * N4H * 4);
    __bf16* hbuf    = (__bf16*)alloc(4L * BB * HH * 2);
    float*  cbuf    = (float*) alloc(4L * BB * HH * 4);
    float*  zbuf    = (float*) alloc(4L * BB * N4H * 4);
    __bf16* W1t     = (__bf16*)alloc((long)FF * N4H * 2);
    __bf16* W2t     = (__bf16*)alloc((long)FF * FF * 2);
    __bf16* W3t     = (__bf16*)alloc((long)FF * FF * 2);
    __bf16* rnnbf   = (__bf16*)alloc((long)BB * N4H * 2);
    __bf16* a1bf    = (__bf16*)alloc((long)BB * FF * 2);
    __bf16* a2bf    = (__bf16*)alloc((long)BB * FF * 2);
    __bf16* a3bf    = (__bf16*)alloc((long)BB * FF * 2);

    const int TPB = 256;
    // --- setup ---
    {
        long n = 2L * BB * TT * EP;
        k_convert_x<<<(unsigned)((n + TPB - 1) / TPB), TPB, 0, stream>>>(premises, hyps, xp);
    }
    {
        long n = 4L * N4H * KP;
        k_convert_lstm_w<<<(unsigned)((n + TPB - 1) / TPB), TPB, 0, stream>>>(
            W_lstm[0], W_lstm[1], W_lstm[2], W_lstm[3], Wtp);
    }
    k_copy_bias<<<(4 * N4H + TPB - 1) / TPB, TPB, 0, stream>>>(
        b_lstm[0], b_lstm[1], b_lstm[2], b_lstm[3], biasAll);
    k_convert_mlp_w<<<(unsigned)(((long)FF * N4H + TPB - 1) / TPB), TPB, 0, stream>>>(W1, W1t, FF, N4H);
    k_convert_mlp_w<<<(unsigned)(((long)FF * FF + TPB - 1) / TPB), TPB, 0, stream>>>(W2, W2t, FF, FF);
    k_convert_mlp_w<<<(unsigned)(((long)FF * FF + TPB - 1) / TPB), TPB, 0, stream>>>(W3, W3t, FF, FF);
    k_init_hc<<<(4 * BB * HH + TPB - 1) / TPB, TPB, 0, stream>>>(hbuf, cbuf);

    // --- recurrent loop: 4 LSTMs batched per step ---
    // waves = 4 * (256/32) * (2048/32) = 2048  -> 256 blocks of 256 threads
    const int gemm_blocks = (4 * (BB / 32) * (N4H / 32) * 32) / TPB;
    const int cell_blocks = (4 * BB * HH + TPB - 1) / TPB;
    for (int s = 0; s < TT; ++s) {
        lstm_step_gemm<<<gemm_blocks, TPB, 0, stream>>>(xp, Wtp, hbuf, zbuf, s);
        lstm_cell<<<cell_blocks, TPB, 0, stream>>>(zbuf, biasAll, cbuf, hbuf);
    }

    // --- MLP head ---
    k_repack<<<cell_blocks, TPB, 0, stream>>>(cbuf, rnnbf);
    // waves = (256/32) * (N/32); blocks = waves*32/256 = N/32
    mlp_gemm_tanh<<<FF / 32, TPB, 0, stream>>>(rnnbf, W1t, b1, a1bf, FF, N4H);
    mlp_gemm_tanh<<<FF / 32, TPB, 0, stream>>>(a1bf, W2t, b2, a2bf, FF, FF);
    mlp_gemm_tanh<<<FF / 32, TPB, 0, stream>>>(a2bf, W3t, b3, a3bf, FF, FF);
    k_logits<<<(BB * 3 + TPB - 1) / TPB, TPB, 0, stream>>>(a3bf, W4, b4, out);
}
